// WindowAttention3D_17703855194343
// MI455X (gfx1250) — matrix operand order repaired and checked by reading, not compiled, not run on hardware
//
#include <hip/hip_runtime.h>
#include <hip/hip_fp16.h>
#include <math.h>

#define B_ 2
#define N_ 1024
#define C_ 512
#define H_ 16
#define D_ 32
#define PH_ 128
#define GRID_RES 3

typedef _Float16 half_t;
typedef __attribute__((ext_vector_type(16))) _Float16 v16h;
typedef __attribute__((ext_vector_type(8)))  _Float16 v8h;
typedef __attribute__((ext_vector_type(8)))  float    v8f;
typedef __attribute__((ext_vector_type(4)))  float    v4f;

static __device__ inline v8f wmma(v16h a, v16h b, v8f c) {
  return __builtin_amdgcn_wmma_f32_16x16x32_f16(false, a, false, b, (short)0, c, false, false);
}

// A-matrix 16x32 f16 fragment: lane l -> row M=l%16; kb=16*(l>=16);
// halves[0..15]=A[M][kb..kb+15]
static __device__ inline v16h loadA16x32(const half_t* base, int ld, int lane) {
  int M = lane & 15;
  int kb = (lane >> 4) * 16;
  const half_t* row = base + M * ld;
  v8h lo = *(const v8h*)(row + kb);
  v8h hi = *(const v8h*)(row + kb + 8);
  v16h a;
#pragma unroll
  for (int i = 0; i < 8; ++i) { a[i] = lo[i]; a[i + 8] = hi[i]; }
  return a;
}

// B-matrix 32x16 f16 fragment, element (k, col) at base[col*ld + k].
// lane l -> col=l%16, K=16*(l>=16)..+15 (contiguous 32B load).
static __device__ inline v16h loadB32x16(const half_t* base, int ld, int lane) {
  int col = lane & 15;
  int kb = (lane >> 4) * 16;
  return *(const v16h*)(base + col * ld + kb);
}

// ---------------------------------------------------------------------------
// Kernel 1: coordinate min/max + window partition ids
// ---------------------------------------------------------------------------
__global__ void prep_kernel(const float* __restrict__ coords, int* __restrict__ wid) {
  int b = blockIdx.x;
  int t = threadIdx.x; // 256 threads
  float mn[3] = {1e30f, 1e30f, 1e30f};
  float mx[3] = {-1e30f, -1e30f, -1e30f};
  for (int n = t; n < N_; n += 256) {
    const float* c = coords + ((size_t)b * N_ + n) * 3;
#pragma unroll
    for (int d = 0; d < 3; ++d) {
      float v = c[d];
      mn[d] = fminf(mn[d], v);
      mx[d] = fmaxf(mx[d], v);
    }
  }
  for (int msk = 1; msk < 32; msk <<= 1) {
#pragma unroll
    for (int d = 0; d < 3; ++d) {
      mn[d] = fminf(mn[d], __shfl_xor(mn[d], msk, 32));
      mx[d] = fmaxf(mx[d], __shfl_xor(mx[d], msk, 32));
    }
  }
  __shared__ float smn[8][3], smx[8][3];
  __shared__ float bmn[3], bmx[3];
  int w = t >> 5, l = t & 31;
  if (l == 0) {
#pragma unroll
    for (int d = 0; d < 3; ++d) { smn[w][d] = mn[d]; smx[w][d] = mx[d]; }
  }
  __syncthreads();
  if (t == 0) {
#pragma unroll
    for (int d = 0; d < 3; ++d) {
      float a = 1e30f, bb = -1e30f;
      for (int i = 0; i < 8; ++i) { a = fminf(a, smn[i][d]); bb = fmaxf(bb, smx[i][d]); }
      bmn[d] = a; bmx[d] = bb;
    }
  }
  __syncthreads();
  for (int n = t; n < N_; n += 256) {
    const float* c = coords + ((size_t)b * N_ + n) * 3;
    int bin[3];
#pragma unroll
    for (int d = 0; d < 3; ++d) {
      float r = bmx[d] - bmn[d];
      if (r < 1e-6f) r = 1.0f;
      float cn = (c[d] - bmn[d]) / r;
      int bi = (int)(cn * (float)GRID_RES);
      bi = bi < 0 ? 0 : (bi > GRID_RES - 1 ? GRID_RES - 1 : bi);
      bin[d] = bi;
    }
    wid[b * N_ + n] = bin[0] * 9 + bin[1] * 3 + bin[2];
  }
}

// ---------------------------------------------------------------------------
// Kernel 2: QKV projection GEMM (f32 in -> f16 staged -> WMMA).
// q,k stored [b][h][n][d]; V stored TRANSPOSED [b][h][d][n] so the attention
// P@V B-fragment is one contiguous 32B load.
// ---------------------------------------------------------------------------
__global__ void __launch_bounds__(256) qkv_kernel(
    const float* __restrict__ x, const float* __restrict__ wq,
    const float* __restrict__ bq,
    half_t* __restrict__ qh, half_t* __restrict__ kh, half_t* __restrict__ vt) {
  int b = blockIdx.z;
  int n0 = blockIdx.x * 128;
  int c0 = blockIdx.y * 64;
  int t = threadIdx.x, lane = t & 31, wv = t >> 5;
  __shared__ alignas(64) half_t xs[128 * 32];
  __shared__ alignas(64) half_t wsh[64 * 32];
  v8f acc[4] = {};
  for (int k0 = 0; k0 < C_; k0 += 32) {
    {
      int r = t >> 1, cb = (t & 1) * 16;
      const float* src = x + ((size_t)(b * N_ + n0 + r)) * C_ + k0 + cb;
      half_t* dst = xs + r * 32 + cb;
#pragma unroll
      for (int i = 0; i < 16; i += 4) {
        v4f f = *(const v4f*)(src + i);
        dst[i] = (half_t)f[0]; dst[i + 1] = (half_t)f[1];
        dst[i + 2] = (half_t)f[2]; dst[i + 3] = (half_t)f[3];
      }
      if (k0 + 32 < C_) __builtin_prefetch(src + 32, 0, 1);
    }
    {
      int r = t >> 2, cb = (t & 3) * 8;
      const float* src = wq + ((size_t)(c0 + r)) * C_ + k0 + cb;
      half_t* dst = wsh + r * 32 + cb;
#pragma unroll
      for (int i = 0; i < 8; i += 4) {
        v4f f = *(const v4f*)(src + i);
        dst[i] = (half_t)f[0]; dst[i + 1] = (half_t)f[1];
        dst[i + 2] = (half_t)f[2]; dst[i + 3] = (half_t)f[3];
      }
    }
    __syncthreads();
    v16h a = loadA16x32(xs + wv * 16 * 32, 32, lane);
#pragma unroll
    for (int nt = 0; nt < 4; ++nt) {
      v16h bmat = loadB32x16(wsh + nt * 16 * 32, 32, lane);
      acc[nt] = wmma(a, bmat, acc[nt]);
    }
    __syncthreads();
  }
  int hl = lane >> 4, jc = lane & 15;
#pragma unroll
  for (int nt = 0; nt < 4; ++nt) {
    int col = c0 + nt * 16 + jc;          // 0..1535
    float bv = bq[col];
    int sec = col / C_;                   // 0=q 1=k 2=v
    int cl = col % C_;
    int hh = cl >> 5, d = cl & 31;
#pragma unroll
    for (int r = 0; r < 8; ++r) {
      int n = n0 + wv * 16 + r + 8 * hl;
      half_t val = (half_t)(acc[nt][r] + bv);
      if (sec == 0)
        qh[(((size_t)(b * H_ + hh)) * N_ + n) * D_ + d] = val;
      else if (sec == 1)
        kh[(((size_t)(b * H_ + hh)) * N_ + n) * D_ + d] = val;
      else
        vt[(((size_t)(b * H_ + hh)) * D_ + d) * N_ + n] = val;   // transposed
    }
  }
}

// ---------------------------------------------------------------------------
// Kernel 3: fused pos-bias MLP (window-compacted) + masked flash attention.
// block: 512 thr (16 waves), wave w = head w. One block per (b, 16-row tile).
// ---------------------------------------------------------------------------
__global__ void __launch_bounds__(512) attn_kernel(
    const float* __restrict__ coords,
    const float* __restrict__ w1, const float* __restrict__ b1,
    const float* __restrict__ w2, const float* __restrict__ b2,
    const half_t* __restrict__ qh, const half_t* __restrict__ kh,
    const half_t* __restrict__ vt, const int* __restrict__ wid,
    half_t* __restrict__ ao) {
  int b = blockIdx.y;
  int i0 = blockIdx.x * 16;
  int t = threadIdx.x, lane = t & 31, w = t >> 5;   // w = head
  int hl = lane >> 4, jc = lane & 15;

  __shared__ alignas(64) half_t hidLds[128 * 128]; // 32 KB: compact hid / P staging
  __shared__ alignas(64) half_t biasLds[512 * 16]; // 16 KB: (pair x head) bias
  __shared__ int wjs[32];
  __shared__ unsigned short list[512];
  __shared__ int wcnt[16], wbase[16], nactS;

  // preload pos_w2 B-fragments (4 chunks of K=32) and pos_b2 lane value
  v16h w2f[4];
  {
    int kb = 16 * hl;
#pragma unroll
    for (int cc = 0; cc < 4; ++cc) {
      v16h f;
#pragma unroll
      for (int tt = 0; tt < 16; ++tt)
        f[tt] = (half_t)w2[jc * PH_ + cc * 32 + kb + tt];
      w2f[cc] = f;
    }
  }
  float pb2 = b2[jc];

  // q A-fragment for this head / row tile
  v16h qf;
  {
    int M = lane & 15, kb = 16 * hl;
    const half_t* qp = qh + (((size_t)(b * H_ + w)) * N_ + i0 + M) * D_;
    v8h lo = *(const v8h*)(qp + kb);
    v8h hi = *(const v8h*)(qp + kb + 8);
#pragma unroll
    for (int i = 0; i < 8; ++i) { qf[i] = lo[i]; qf[i + 8] = hi[i]; }
  }
  int widi[8];
#pragma unroll
  for (int r = 0; r < 8; ++r) widi[r] = wid[b * N_ + i0 + r + 8 * hl];
  // window id of this thread's pair-row (pair p = t -> row il = t>>5)
  int wiPair = wid[b * N_ + i0 + (t >> 5)];
  int jlPair = t & 31;

  float m[8], ll[8];
  v8f o0 = {}, o1 = {};
#pragma unroll
  for (int r = 0; r < 8; ++r) { m[r] = -3.0e38f; ll[r] = 0.f; }
  const float scale = 0.17677669529663687f;  // 32^-0.5

  for (int jt = 0; jt < N_ / 32; ++jt) {
    int j0 = jt * 32;
    __syncthreads();   // protects hidLds (P regions) + wjs + list from prev iter
    if (t < 32) wjs[t] = wid[b * N_ + j0 + t];
    __syncthreads();

    // ---- compact worklist of same-window pairs (mask kills ~96%) ----
    bool samew = (wiPair == wjs[jlPair]);
    unsigned msk32 = (unsigned)__ballot(samew);
    if (lane == 0) wcnt[w] = __popc(msk32);
    __syncthreads();
    if (t == 0) {
      int s = 0;
#pragma unroll
      for (int i = 0; i < 16; ++i) { wbase[i] = s; s += wcnt[i]; }
      nactS = s;
    }
    __syncthreads();
    int nact = nactS;
    if (nact == 0) continue;  // uniform: fully masked tile contributes exactly 0
    if (samew) {
      int idx = wbase[w] + __popc(msk32 & ((1u << lane) - 1u));
      list[idx] = (unsigned short)t;
    }
    __syncthreads();

    // ---- hid (full K=128) for active pairs only, then bias WMMA ----
    for (int base = 0; base < nact; base += 128) {
      int nb = nact - base; if (nb > 128) nb = 128;
      {
        int e = t & 127, q = t >> 7;      // 4 threads per compact pair
        if (e < nb) {
          int p = list[base + e];
          int pil = p >> 5, pjl = p & 31;
          const float* cI = coords + ((size_t)b * N_ + i0 + pil) * 3;
          const float* cJ = coords + ((size_t)b * N_ + j0 + pjl) * 3;
          float ox = cI[0] - cJ[0], oy = cI[1] - cJ[1], oz = cI[2] - cJ[2];
          half_t* hrow = hidLds + e * 128 + q * 32;
#pragma unroll 8
          for (int k = 0; k < 32; ++k) {
            int kk = q * 32 + k;
            float s = ox * w1[kk * 3 + 0] + oy * w1[kk * 3 + 1] +
                      oz * w1[kk * 3 + 2] + b1[kk];
            hrow[k] = (half_t)(0.5f * s * (1.0f + erff(s * 0.70710678118654752f)));
          }
        }
      }
      __syncthreads();
      if (w < 8 && w * 16 < nb) {   // 16-compact-row tiles x 16 heads
        v8f vb = {};
#pragma unroll
        for (int cc = 0; cc < 4; ++cc) {
          v16h a = loadA16x32(hidLds + (w * 16) * 128 + cc * 32, 128, lane);
          vb = wmma(a, w2f[cc], vb);
        }
#pragma unroll
        for (int r = 0; r < 8; ++r) {
          int row = w * 16 + r + 8 * hl;
          if (row < nb) {
            int pair = list[base + row];
            biasLds[pair * 16 + jc] = (half_t)(vb[r] + pb2);
          }
        }
      }
      __syncthreads();
    }

    // ---- S = q @ k^T (one WMMA per 16-col half) ----
    const half_t* kbase = kh + (((size_t)(b * H_ + w)) * N_ + j0) * D_;
    v16h kf0 = loadB32x16(kbase, D_, lane);
    v16h kf1 = loadB32x16(kbase + 16 * D_, D_, lane);
    v8f z = {};
    v8f s0 = wmma(qf, kf0, z);
    v8f s1 = wmma(qf, kf1, z);

    // ---- bias + window mask + online softmax ----
    float p0v[8], p1v[8];
#pragma unroll
    for (int r = 0; r < 8; ++r) {
      int il2 = r + 8 * hl;
      float bb0 = (float)biasLds[(il2 * 32 + jc) * 16 + w];
      float bb1 = (float)biasLds[(il2 * 32 + 16 + jc) * 16 + w];
      float v0 = (widi[r] == wjs[jc]) ? (s0[r] * scale + bb0) : -10000.0f;
      float v1 = (widi[r] == wjs[16 + jc]) ? (s1[r] * scale + bb1) : -10000.0f;
      float rm = fmaxf(v0, v1);
      for (int msk = 1; msk < 16; msk <<= 1) rm = fmaxf(rm, __shfl_xor(rm, msk, 32));
      float mn2 = fmaxf(m[r], rm);
      float alpha = __expf(m[r] - mn2);
      float e0 = __expf(v0 - mn2), e1 = __expf(v1 - mn2);
      float rs = e0 + e1;
      for (int msk = 1; msk < 16; msk <<= 1) rs += __shfl_xor(rs, msk, 32);
      ll[r] = ll[r] * alpha + rs;
      m[r] = mn2;
      o0[r] *= alpha; o1[r] *= alpha;
      p0v[r] = e0; p1v[r] = e1;
    }

    // ---- P (C layout) -> LDS -> A fragment; O += P @ V ----
    half_t* pr = hidLds + (w * 32) * 32;  // private per-wave region
#pragma unroll
    for (int r = 0; r < 8; ++r) {
      int il2 = r + 8 * hl;
      pr[il2 * 32 + jc] = (half_t)p0v[r];
      pr[il2 * 32 + 16 + jc] = (half_t)p1v[r];
    }
    v16h pf = loadA16x32(pr, 32, lane);
    // V is stored transposed [b][h][d][n]: contiguous-in-j B fragments
    const half_t* vbase = vt + (((size_t)(b * H_ + w)) * D_) * N_ + j0;
    v16h vf0 = loadB32x16(vbase, N_, lane);            // cols d = 0..15
    v16h vf1 = loadB32x16(vbase + 16 * N_, N_, lane);  // cols d = 16..31
    o0 = wmma(pf, vf0, o0);
    o1 = wmma(pf, vf1, o1);
  }

  // ---- normalize and store (B,N,C) f16 ----
#pragma unroll
  for (int r = 0; r < 8; ++r) {
    float inv = 1.0f / ll[r];
    int n = i0 + r + 8 * hl;
    int c0 = w * D_ + jc;
    ao[((size_t)b * N_ + n) * C_ + c0] = (half_t)(o0[r] * inv);
    ao[((size_t)b * N_ + n) * C_ + c0 + 16] = (half_t)(o1[r] * inv);
  }
}

// ---------------------------------------------------------------------------
// Kernel 4: output projection GEMM (f16 attn-out @ proj_w^T + proj_b -> f32)
// ---------------------------------------------------------------------------
__global__ void __launch_bounds__(256) proj_kernel(
    const half_t* __restrict__ ao, const float* __restrict__ wp,
    const float* __restrict__ bp, float* __restrict__ out) {
  int row0 = blockIdx.x * 128;  // rows over B*N = 2048
  int c0 = blockIdx.y * 64;
  int t = threadIdx.x, lane = t & 31, wv = t >> 5;
  __shared__ alignas(64) half_t xs[128 * 32];
  __shared__ alignas(64) half_t wsh[64 * 32];
  v8f acc[4] = {};
  for (int k0 = 0; k0 < C_; k0 += 32) {
    {
      int r = t >> 1, cb = (t & 1) * 16;
      const half_t* src = ao + ((size_t)(row0 + r)) * C_ + k0 + cb;
      v8h a = *(const v8h*)src;
      v8h b2v = *(const v8h*)(src + 8);
      half_t* dst = xs + r * 32 + cb;
      *(v8h*)dst = a;
      *(v8h*)(dst + 8) = b2v;
      if (k0 + 32 < C_) __builtin_prefetch(src + 32, 0, 1);
    }
    {
      int r = t >> 2, cb = (t & 3) * 8;
      const float* src = wp + ((size_t)(c0 + r)) * C_ + k0 + cb;
      half_t* dst = wsh + r * 32 + cb;
#pragma unroll
      for (int i = 0; i < 8; i += 4) {
        v4f f = *(const v4f*)(src + i);
        dst[i] = (half_t)f[0]; dst[i + 1] = (half_t)f[1];
        dst[i + 2] = (half_t)f[2]; dst[i + 3] = (half_t)f[3];
      }
    }
    __syncthreads();
    v16h a = loadA16x32(xs + wv * 16 * 32, 32, lane);
#pragma unroll
    for (int nt = 0; nt < 4; ++nt) {
      v16h bmat = loadB32x16(wsh + nt * 16 * 32, 32, lane);
      acc[nt] = wmma(a, bmat, acc[nt]);
    }
    __syncthreads();
  }
  int hl = lane >> 4, jc = lane & 15;
#pragma unroll
  for (int nt = 0; nt < 4; ++nt) {
    int col = c0 + nt * 16 + jc;
    float bv = bp[col];
#pragma unroll
    for (int r = 0; r < 8; ++r) {
      int row = row0 + wv * 16 + r + 8 * hl;
      out[(size_t)row * C_ + col] = acc[nt][r] + bv;
    }
  }
}

// ---------------------------------------------------------------------------
extern "C" void kernel_launch(void* const* d_in, const int* in_sizes, int n_in,
                              void* d_out, int out_size, void* d_ws, size_t ws_size,
                              hipStream_t stream) {
  (void)in_sizes; (void)n_in; (void)out_size; (void)ws_size;
  const float* coords = (const float*)d_in[0];
  const float* x      = (const float*)d_in[1];
  const float* qkv_w  = (const float*)d_in[2];
  const float* qkv_b  = (const float*)d_in[3];
  const float* proj_w = (const float*)d_in[4];
  const float* proj_b = (const float*)d_in[5];
  const float* pos_w1 = (const float*)d_in[6];
  const float* pos_b1 = (const float*)d_in[7];
  const float* pos_w2 = (const float*)d_in[8];
  const float* pos_b2 = (const float*)d_in[9];
  float* out = (float*)d_out;

  char* ws = (char*)d_ws;
  const size_t SZH = (size_t)B_ * H_ * N_ * D_ * sizeof(half_t);      // 2 MB each
  half_t* qh = (half_t*)(ws);
  half_t* kh = (half_t*)(ws + SZH);
  half_t* vt = (half_t*)(ws + 2 * SZH);                               // [b][h][d][n]
  half_t* ao = (half_t*)(ws + 3 * SZH);                               // B*N*C f16 = 2 MB
  int* wid   = (int*)(ws + 3 * SZH + (size_t)B_ * N_ * C_ * sizeof(half_t));

  prep_kernel<<<dim3(B_), 256, 0, stream>>>(coords, wid);
  qkv_kernel<<<dim3(N_ / 128, (3 * C_) / 64, B_), 256, 0, stream>>>(
      x, qkv_w, qkv_b, qh, kh, vt);
  attn_kernel<<<dim3(N_ / 16, B_), 512, 0, stream>>>(
      coords, pos_w1, pos_b1, pos_w2, pos_b2, qh, kh, vt, wid, ao);
  proj_kernel<<<dim3((B_ * N_) / 128, C_ / 64), 256, 0, stream>>>(
      ao, proj_w, proj_b, out);
}
